// SelfAttention_16466904613551
// MI455X (gfx1250) — matrix operand order repaired and checked by reading, not compiled, not run on hardware
//
#include <hip/hip_runtime.h>

// ---------------------------------------------------------------------------
// CDNA5 (gfx1250) causal self-attention: fp32 in/out, bf16 WMMA internally.
// Round 2: 64-key attention iterations (halves per-key softmax/rescale VALU),
//          1/sqrt(dk) folded into the q projection.
// ---------------------------------------------------------------------------

typedef __attribute__((ext_vector_type(16))) __bf16 v16bf;
typedef __attribute__((ext_vector_type(8)))  __bf16 v8bf;
typedef __attribute__((ext_vector_type(8)))  float  v8f;
typedef __attribute__((ext_vector_type(8)))  float  f32x8;
typedef __attribute__((ext_vector_type(16))) float  f32x16;

#define DIM_IN 512
#define DIM_K  256
#define DIM_V  256
#define BATCH  4
#define SEQ    4096

// ds_swizzle XOR within 32 lanes (and_mask=0x1f; mask<16 keeps 16-lane groups)
#define SWZ(v, mask) __int_as_float(__builtin_amdgcn_ds_swizzle(__float_as_int(v), (((mask) << 10) | 0x1f)))

// ===========================================================================
// Kernel 1: q = (xWq+bq)/16, k = xWk+bk (stored transposed), v = xWv+bv.
// grid = (B*S/64, 3, DIM_K/64), block = 128 (4 waves).
// Wave w computes a 16-row x 64-col tile via 4 WMMA accumulators.
// ===========================================================================
__global__ __launch_bounds__(128) void qkv_proj_kernel(
    const float* __restrict__ x,
    const float* __restrict__ Wq, const float* __restrict__ bq,
    const float* __restrict__ Wk, const float* __restrict__ bk,
    const float* __restrict__ Wv, const float* __restrict__ bv,
    __bf16* __restrict__ qb, __bf16* __restrict__ kt, __bf16* __restrict__ vb)
{
  const int wave = threadIdx.x >> 5;
  const int lane = threadIdx.x & 31;
  const int half = lane >> 4;
  const int mrow = lane & 15;
  const int gr   = blockIdx.x * 64 + wave * 16;   // first global row of this wave's strip
  const int mat  = blockIdx.y;                    // 0=q 1=k 2=v
  const int colbase = blockIdx.z * 64;

  const float* W    = (mat == 0) ? Wq : (mat == 1) ? Wk : Wv;
  const float* bias = (mat == 0) ? bq : (mat == 1) ? bk : bv;
  const float  oscale = (mat == 0) ? 0.0625f : 1.0f;   // fold 1/sqrt(256) into q

  v8f acc[4];
  #pragma unroll
  for (int t = 0; t < 4; ++t)
    #pragma unroll
    for (int j = 0; j < 8; ++j) acc[t][j] = 0.0f;

  // A operand: row M = mrow, K runs at half*8 and 16+half*8 (ISA A-layout).
  const float* xrow = x + (size_t)(gr + mrow) * DIM_IN;
  #pragma unroll
  for (int kc = 0; kc < DIM_IN; kc += 32) {
    f32x8 lo = *(const f32x8*)(xrow + kc + half * 8);
    f32x8 hi = *(const f32x8*)(xrow + kc + 16 + half * 8);
    v16bf a;
    #pragma unroll
    for (int e = 0; e < 8; ++e) { a[e] = (__bf16)lo[e]; a[8 + e] = (__bf16)hi[e]; }

    // B operand: lane holds column N = l&15, K runs at half*8 and 16+half*8.
    const float* wcol = W + (size_t)kc * DIM_K + colbase + mrow;
    #pragma unroll
    for (int t = 0; t < 4; ++t) {
      v16bf bmat;
      #pragma unroll
      for (int e = 0; e < 16; ++e) {
        const int kk = (e < 8 ? e : e + 8) + half * 8;
        bmat[e] = (__bf16)wcol[(size_t)kk * DIM_K + t * 16];
      }
      acc[t] = __builtin_amdgcn_wmma_f32_16x16x32_bf16(
          false, a, false, bmat, (short)0, acc[t], false, false);
    }
  }

  // Epilogue: bias, scale, convert bf16, write (k transposed: kt[b][col][s]).
  #pragma unroll
  for (int t = 0; t < 4; ++t) {
    const int col = colbase + t * 16 + mrow;      // C/D layout: N = lane & 15
    const float bval = bias[col];
    #pragma unroll
    for (int j = 0; j < 8; ++j) {
      const int row = gr + j + 8 * half;          // C/D layout: M = j + 8*(lane>>4)
      const float val = (acc[t][j] + bval) * oscale;
      if (mat == 0) {
        qb[(size_t)row * DIM_K + col] = (__bf16)val;
      } else if (mat == 1) {
        const int bb = row >> 12;                 // SEQ = 4096
        const int s  = row & 4095;
        kt[((size_t)bb * DIM_K + col) * SEQ + s] = (__bf16)val;
      } else {
        vb[(size_t)row * DIM_V + col] = (__bf16)val;
      }
    }
  }
}

// ===========================================================================
// Kernel 2: flash attention with online softmax, 64 keys per iteration.
// grid = (S/64, B), block = 128 (4 waves). Wave w owns q rows [q0+16w, +16).
// Per iteration: 32 WMMA scores (4 tiles x 8 dk-chunks, in 2-tile sub-loops),
// fp32 online softmax, P relayout C->A via per-wave LDS, 32 WMMA for P@V.
// ===========================================================================
__global__ __launch_bounds__(128) void attn_kernel(
    const __bf16* __restrict__ qb, const __bf16* __restrict__ kt,
    const __bf16* __restrict__ vb, float* __restrict__ out)
{
  __shared__ alignas(32) __bf16 Qs[64 * DIM_K];    // 32 KB: this block's Q tile
  __shared__ alignas(32) __bf16 Ps[4][16 * 64];    // 8 KB : per-wave P staging

  const int b  = blockIdx.y;
  const int q0 = blockIdx.x * 64;

  // Cooperative Q load (bf16, row-major, already pre-scaled by 1/16) into LDS.
  const __bf16* qsrc = qb + ((size_t)b * SEQ + q0) * DIM_K;
  for (int i = threadIdx.x * 8; i < 64 * DIM_K; i += 128 * 8)
    *(v8bf*)(Qs + i) = *(const v8bf*)(qsrc + i);
  __syncthreads();

  const int wave = threadIdx.x >> 5;
  const int lane = threadIdx.x & 31;
  const int half = lane >> 4;
  const int mrow = lane & 15;
  const int lr0  = wave * 16;       // local first row
  const int qr0  = q0 + lr0;        // global first row

  float m_i[8], l_i[8];
  v8f O[16];
  #pragma unroll
  for (int j = 0; j < 8; ++j) { m_i[j] = -3.0e38f; l_i[j] = 0.0f; }
  #pragma unroll
  for (int t = 0; t < 16; ++t)
    #pragma unroll
    for (int j = 0; j < 8; ++j) O[t][j] = 0.0f;

  const __bf16* Kbase = kt + (size_t)b * DIM_K * SEQ;
  const __bf16* Vbase = vb + (size_t)b * SEQ * DIM_V;

  for (int n0 = 0; n0 < q0 + 64; n0 += 64) {
    // ---- scores for 64 keys: 4 tiles of 16, computed as 2 pairs ----------
    v8f sc[4];
    #pragma unroll
    for (int p = 0; p < 2; ++p) {
      v8f sa, sb;
      #pragma unroll
      for (int j = 0; j < 8; ++j) { sa[j] = 0.0f; sb[j] = 0.0f; }
      #pragma unroll
      for (int kc = 0; kc < DIM_K; kc += 32) {
        const __bf16* qp = Qs + (size_t)(lr0 + mrow) * DIM_K + kc + half * 8;
        v8bf qlo = *(const v8bf*)qp;
        v8bf qhi = *(const v8bf*)(qp + 16);
        v16bf a;
        #pragma unroll
        for (int e = 0; e < 8; ++e) { a[e] = qlo[e]; a[8 + e] = qhi[e]; }

        const __bf16* kcol = Kbase + (size_t)kc * SEQ + n0 + p * 32 + mrow; // key column
        v16bf b0, b1;
        #pragma unroll
        for (int e = 0; e < 16; ++e) {
          const int kk = (e < 8 ? e : e + 8) + half * 8;
          b0[e] = kcol[(size_t)kk * SEQ];
          b1[e] = kcol[(size_t)kk * SEQ + 16];
        }
        sa = __builtin_amdgcn_wmma_f32_16x16x32_bf16(false, a, false, b0, (short)0, sa, false, false);
        sb = __builtin_amdgcn_wmma_f32_16x16x32_bf16(false, a, false, b1, (short)0, sb, false, false);
      }
      sc[2 * p]     = sa;
      sc[2 * p + 1] = sb;
    }

    // ---- causal mask + online softmax (fp32; scores pre-scaled via q) ----
    #pragma unroll
    for (int j = 0; j < 8; ++j) {
      const int rowg = qr0 + j + 8 * half;
      float v[4];
      #pragma unroll
      for (int t = 0; t < 4; ++t) {
        const int col = n0 + t * 16 + mrow;
        v[t] = (col <= rowg) ? sc[t][j] : -3.0e38f;
      }
      float mc = fmaxf(fmaxf(v[0], v[1]), fmaxf(v[2], v[3]));
      mc = fmaxf(mc, SWZ(mc, 1));
      mc = fmaxf(mc, SWZ(mc, 2));
      mc = fmaxf(mc, SWZ(mc, 4));
      mc = fmaxf(mc, SWZ(mc, 8));
      const float mn    = fmaxf(m_i[j], mc);
      const float alpha = __expf(m_i[j] - mn);
      float p[4], rs = 0.0f;
      #pragma unroll
      for (int t = 0; t < 4; ++t) { p[t] = __expf(v[t] - mn); rs += p[t]; }
      rs += SWZ(rs, 1);
      rs += SWZ(rs, 2);
      rs += SWZ(rs, 4);
      rs += SWZ(rs, 8);
      l_i[j] = l_i[j] * alpha + rs;
      m_i[j] = mn;
      #pragma unroll
      for (int t = 0; t < 16; ++t) O[t][j] *= alpha;
      // stash P (C layout -> row-major 16x64 tile in LDS)
      #pragma unroll
      for (int t = 0; t < 4; ++t)
        Ps[wave][(j + 8 * half) * 64 + t * 16 + mrow] = (__bf16)p[t];
    }
    __syncthreads();   // identical trip counts for all waves; orders DS ops

    // ---- O += P @ V over two 32-key sub-blocks -----------------------------
    #pragma unroll
    for (int p = 0; p < 2; ++p) {
      // reload P in A layout (lane row = mrow, K runs at half*8 / 16+half*8)
      const __bf16* pp = &Ps[wave][mrow * 64 + p * 32 + half * 8];
      v8bf plo = *(const v8bf*)pp;
      v8bf phi = *(const v8bf*)(pp + 16);
      v16bf pa;
      #pragma unroll
      for (int e = 0; e < 8; ++e) { pa[e] = plo[e]; pa[8 + e] = phi[e]; }

      const int kr = n0 + p * 32;
      const __bf16* vrow = Vbase + (size_t)kr * DIM_V + mrow;
      __builtin_prefetch(vrow + 64 * DIM_V, 0, 1);         // next key block (L2)
      #pragma unroll
      for (int t = 0; t < 16; ++t) {
        v16bf vmat;
        #pragma unroll
        for (int e = 0; e < 16; ++e) {
          const int kk = (e < 8 ? e : e + 8) + half * 8;
          vmat[e] = vrow[(size_t)kk * DIM_V + t * 16];
        }
        O[t] = __builtin_amdgcn_wmma_f32_16x16x32_bf16(
            false, pa, false, vmat, (short)0, O[t], false, false);
      }
    }
  }

  // ---- normalize and write fp32 output ----
  #pragma unroll
  for (int j = 0; j < 8; ++j) {
    const float inv = 1.0f / l_i[j];
    const size_t rowoff = ((size_t)b * SEQ + qr0 + j + 8 * half) * DIM_V;
    #pragma unroll
    for (int t = 0; t < 16; ++t)
      out[rowoff + t * 16 + mrow] = O[t][j] * inv;
  }
}

// ===========================================================================
extern "C" void kernel_launch(void* const* d_in, const int* in_sizes, int n_in,
                              void* d_out, int out_size, void* d_ws, size_t ws_size,
                              hipStream_t stream)
{
  const float* x  = (const float*)d_in[0];
  const float* Wq = (const float*)d_in[1];
  const float* bq = (const float*)d_in[2];
  const float* Wk = (const float*)d_in[3];
  const float* bk = (const float*)d_in[4];
  const float* Wv = (const float*)d_in[5];
  const float* bv = (const float*)d_in[6];
  float* out = (float*)d_out;

  // Workspace: bf16 q [B,S,dk], k^T [B,dk,S], v [B,S,dv]  (8 MB each, 24 MB total)
  __bf16* qb = (__bf16*)d_ws;
  __bf16* kt = qb + (size_t)BATCH * SEQ * DIM_K;
  __bf16* vb = kt + (size_t)BATCH * SEQ * DIM_K;

  qkv_proj_kernel<<<dim3(BATCH * SEQ / 64, 3, DIM_K / 64), 128, 0, stream>>>(
      x, Wq, bq, Wk, bk, Wv, bv, qb, kt, vb);
  attn_kernel<<<dim3(SEQ / 64, BATCH), 128, 0, stream>>>(qb, kt, vb, out);
}
